// AnomalyAttention_46093589020743
// MI455X (gfx1250) — matrix operand order repaired and checked by reading, not compiled, not run on hardware
//
#include <hip/hip_runtime.h>
#include <math.h>

typedef __attribute__((ext_vector_type(16))) _Float16 v16h;
typedef __attribute__((ext_vector_type(8)))  float    v8f;

#define B_    32
#define L_    512
#define H_    8
#define E_    64
#define NT    (L_/16)   // 32 column tiles of the 512-wide score row
#define WAVES 4

__global__ __launch_bounds__(WAVES*32)
void anomaly_attn_kernel(const float* __restrict__ Q,
                         const float* __restrict__ K,
                         const float* __restrict__ V,
                         const float* __restrict__ Sig,
                         float* __restrict__ outV,
                         float* __restrict__ outSeries,
                         float* __restrict__ outPrior,
                         float* __restrict__ outSigma)
{
    // per-wave transposed score tile: sc[col][m], f16. 4*512*16*2 = 64KB
    __shared__ _Float16 sc[WAVES][L_*16];

    const int lane   = threadIdx.x & 31;
    const int wave   = threadIdx.x >> 5;
    const int tileId = blockIdx.x * WAVES + wave;     // 0..8191
    const int b      = tileId >> 8;                   // / (H_*NT)
    const int h      = (tileId >> 5) & (H_-1);
    const int tile   = tileId & (NT-1);
    const int row0   = tile * 16;

    const int n16    = lane & 15;
    const int khalf  = lane >> 4;

    _Float16* S = sc[wave];

    // ---------------- A fragments: Q tile 16x64 as two 16x32 f16 ----------------
    v16h aq[2];
    {
        const long qbase = ((long)(b*L_ + row0 + n16)*H_ + h)*E_;
        #pragma unroll
        for (int ks = 0; ks < 2; ++ks) {
            #pragma unroll
            for (int j = 0; j < 16; ++j) {
                const int kk = ks*32 + j + 16*khalf;             // A layout: K = j + 16*khalf
                aq[ks][j] = (_Float16)Q[qbase + kk];
            }
        }
    }

    float runmax[8];
    #pragma unroll
    for (int r = 0; r < 8; ++r) runmax[r] = -INFINITY;

    // ---------------- pass 1: scores = Q K^T (mask + scale), f16 -> LDS ----------------
    const float scale = 0.125f;   // 1/sqrt(64)
    for (int nt = 0; nt < NT; ++nt) {
        v8f acc = {};
        #pragma unroll
        for (int ks = 0; ks < 2; ++ks) {
            v16h bk;
            const int  scol  = nt*16 + n16;
            const long kbase = ((long)(b*L_ + scol)*H_ + h)*E_ + ks*32 + 16*khalf;
            #pragma unroll
            for (int j = 0; j < 16; ++j)                  // B layout (contiguous, same as A): K = j + 16*khalf
                bk[j] = (_Float16)K[kbase + j];
            acc = __builtin_amdgcn_wmma_f32_16x16x32_f16(false, aq[ks], false, bk,
                                                         (short)0, acc, false, false);
        }
        const int col = nt*16 + n16;
        #pragma unroll
        for (int r = 0; r < 8; ++r) {
            const int m   = r + 8*khalf;                  // C layout: VGPR r -> M = r + 8*khalf
            const int row = row0 + m;
            float v = (col > row) ? -INFINITY : acc[r]*scale;
            runmax[r] = fmaxf(runmax[r], v);
            S[col*16 + m] = (_Float16)v;
        }
    }

    // ---------------- row max across the 16-lane group ----------------
    #pragma unroll
    for (int r = 0; r < 8; ++r) {
        float m = runmax[r];
        m = fmaxf(m, __shfl_xor(m, 1, 32));
        m = fmaxf(m, __shfl_xor(m, 2, 32));
        m = fmaxf(m, __shfl_xor(m, 4, 32));
        m = fmaxf(m, __shfl_xor(m, 8, 32));
        runmax[r] = m;
    }

    // ---------------- pass 2: exponentiate in place, accumulate row sums ----------------
    float runsum[8];
    #pragma unroll
    for (int r = 0; r < 8; ++r) runsum[r] = 0.f;
    for (int nt = 0; nt < NT; ++nt) {
        const int col = nt*16 + n16;
        #pragma unroll
        for (int r = 0; r < 8; ++r) {
            const int m = r + 8*khalf;
            float x = (float)S[col*16 + m];
            float e = __expf(x - runmax[r]);              // masked (-inf) -> 0
            runsum[r] += e;
            S[col*16 + m] = (_Float16)e;
        }
    }
    #pragma unroll
    for (int r = 0; r < 8; ++r) {
        float s = runsum[r];
        s += __shfl_xor(s, 1, 32);
        s += __shfl_xor(s, 2, 32);
        s += __shfl_xor(s, 4, 32);
        s += __shfl_xor(s, 8, 32);
        runsum[r] = 1.0f / s;                             // keep reciprocal
    }

    // ---------------- sigma chain per row ----------------
    float sigv[8], coef[8], ninv[8];
    #pragma unroll
    for (int r = 0; r < 8; ++r) {
        const int row = row0 + r + 8*khalf;
        float s  = Sig[(b*L_ + row)*H_ + h];
        float sg = 1.0f/(1.0f + __expf(-5.0f*s)) + 1e-5f; // sigmoid(5s)+1e-5
        float sv = __expf(1.0986122886681098f*sg) - 1.0f; // 3^sg - 1
        sigv[r] = sv;
        coef[r] = 0.3989422804014327f / sv;               // 1/(sqrt(2pi)*sig)
        ninv[r] = -0.5f/(sv*sv);
    }

    // ---------------- pass 3: normalize, stream series/prior/sigma, P -> LDS ----------------
    const long   phbase = ((long)(b*H_ + h)) * (long)L_ * L_;
    float* __restrict__ serP = outSeries + phbase;
    float* __restrict__ priP = outPrior  + phbase;
    float* __restrict__ sgmP = outSigma  + phbase;

    for (int nt = 0; nt < NT; ++nt) {
        const int col = nt*16 + n16;
        #pragma unroll
        for (int r = 0; r < 8; ++r) {
            const int m   = r + 8*khalf;
            const int row = row0 + m;
            float e = (float)S[col*16 + m];
            float p = e * runsum[r];
            S[col*16 + m] = (_Float16)p;
            const long idx = (long)row*L_ + col;
            __builtin_nontemporal_store(p, serP + idx);
            float dist = (float)(row - col);
            float pr   = coef[r] * __expf(dist*dist*ninv[r]);
            __builtin_nontemporal_store(pr, priP + idx);
            __builtin_nontemporal_store(sigv[r], sgmP + idx);
        }
    }

    __syncthreads();   // cross-lane LDS dependency before GEMM2 A-fragment gathers

    // ---------------- GEMM2: out[16x64] = P[16x512] * V[512x64] ----------------
    v8f oacc[4];
    #pragma unroll
    for (int t = 0; t < 4; ++t) oacc[t] = (v8f){};

    for (int ks = 0; ks < 16; ++ks) {                     // K = 512 in steps of 32
        v16h ap;
        #pragma unroll
        for (int j = 0; j < 16; ++j) {
            const int scol = ks*32 + j + 16*khalf;             // A layout K index: j + 16*khalf
            ap[j] = S[scol*16 + n16];                          // row M = lane%16
        }
        #pragma unroll
        for (int nt2 = 0; nt2 < 4; ++nt2) {
            v16h bv;
            const int  d     = nt2*16 + n16;
            const long vbase = ((long)(b*L_ + ks*32 + 16*khalf)*H_ + h)*E_ + d;
            #pragma unroll
            for (int j = 0; j < 16; ++j)                   // B layout: K = j + 16*khalf
                bv[j] = (_Float16)V[vbase + (long)j*(H_*E_)];
            oacc[nt2] = __builtin_amdgcn_wmma_f32_16x16x32_f16(false, ap, false, bv,
                                                               (short)0, oacc[nt2], false, false);
        }
    }

    // ---------------- write V output [B,L,H,E] ----------------
    #pragma unroll
    for (int nt2 = 0; nt2 < 4; ++nt2) {
        const int d = nt2*16 + n16;
        #pragma unroll
        for (int r = 0; r < 8; ++r) {
            const int row = row0 + r + 8*khalf;
            outV[((long)(b*L_ + row)*H_ + h)*E_ + d] = oacc[nt2][r];
        }
    }
}

extern "C" void kernel_launch(void* const* d_in, const int* in_sizes, int n_in,
                              void* d_out, int out_size, void* d_ws, size_t ws_size,
                              hipStream_t stream) {
    const float* Q   = (const float*)d_in[0];   // [B,L,H,E]
    const float* K   = (const float*)d_in[1];   // [B,L,H,E]
    const float* V   = (const float*)d_in[2];   // [B,L,H,E]
    const float* Sig = (const float*)d_in[3];   // [B,L,H]
    // d_in[4] = attention_mask: causal triu, recomputed in-kernel.

    float* out       = (float*)d_out;
    float* outV      = out;                                   // B*L*H*E   = 8388608
    float* outSeries = out + (long)B_*L_*H_*E_;               // B*H*L*L   = 67108864
    float* outPrior  = outSeries + (long)B_*H_*L_*L_;
    float* outSigma  = outPrior  + (long)B_*H_*L_*L_;

    const int tiles  = B_*H_*NT;          // 8192 waves
    const int blocks = tiles / WAVES;     // 2048
    anomaly_attn_kernel<<<blocks, WAVES*32, 0, stream>>>(Q, K, V, Sig,
                                                         outV, outSeries, outPrior, outSigma);
}
